// Decoder_Seq2seq_GRU_2_14851996910082
// MI455X (gfx1250) — matrix operand order repaired and checked by reading, not compiled, not run on hardware
//
#include <hip/hip_runtime.h>
#include <hip/hip_bf16.h>

// Problem constants (reference: B,T,H,E,V,ENC = 2048,48,512,256,128,1024)
#define Bc   2048
#define Tc   48
#define Hc   512
#define Ec   256
#define Vc   128
#define ENCc 1024
#define H3c  1536   // 3*H

typedef __bf16 bf16x8 __attribute__((ext_vector_type(8)));
typedef __bf16 v16bf  __attribute__((ext_vector_type(16)));
typedef float  v8f    __attribute__((ext_vector_type(8)));

__device__ __forceinline__ v16bf join16(bf16x8 lo, bf16x8 hi) {
  return __builtin_shufflevector(lo, hi, 0,1,2,3,4,5,6,7,8,9,10,11,12,13,14,15);
}

// A fragment, 16x32 bf16 (ISA 7.12.2): lane group 0-15 holds K=[kh..kh+7] in v0-3
// and K=[16+kh..23+kh] in v4-7 with kh = (lane>=16 ? 8 : 0). p points at A[m][k0+kh].
__device__ __forceinline__ v16bf load_a_frag(const __bf16* p) {
  bf16x8 lo = *(const bf16x8*)(p);
  bf16x8 hi = *(const bf16x8*)(p + 16);
  return join16(lo, hi);
}
// B fragment, 32x16 bf16 (same K order as A): lane holds K=[kh..kh+7] and
// K=[16+kh..23+kh], kh = (lane>=16 ? 8 : 0). p points at Bt[n][k0+kh].
__device__ __forceinline__ v16bf load_b_frag(const __bf16* p) {
  bf16x8 lo = *(const bf16x8*)(p);
  bf16x8 hi = *(const bf16x8*)(p + 16);
  return join16(lo, hi);
}

// C[M,N](f32, +bias) = A[M,K](bf16 row-major) * Bt[N,K](bf16, K-major == B^T)
// Block: 256 threads = 8 waves. Wave (wid&3, wid>>2) computes 16(M) x 64(N):
// one A fragment feeds 4 WMMAs per K-step. Block tile: 64(M) x 128(N).
__global__ __launch_bounds__(256)
void gemm_bf16_nt(const __bf16* __restrict__ A, const __bf16* __restrict__ Bt,
                  const float* __restrict__ bias, float* __restrict__ C,
                  int M, int N, int K)
{
  const int lane   = threadIdx.x & 31;
  const int wid    = threadIdx.x >> 5;
  const int laneLo = lane & 15;
  const int hi     = lane >> 4;                 // 0 or 1 (half-wave)
  const int mTile  = blockIdx.y * 64  + (wid & 3) * 16;
  const int nBase  = blockIdx.x * 128 + (wid >> 2) * 64;

  const __bf16* aPtr = A  + (size_t)(mTile + laneLo) * K + (hi << 3);   // +8 for upper half
  const __bf16* bPtr = Bt + (size_t)(nBase + laneLo) * K + (hi << 3);   // +8 for upper half
  const size_t bStride = (size_t)16 * K;

  v8f acc[4] = { v8f{}, v8f{}, v8f{}, v8f{} };

  for (int k = 0; k < K; k += 32) {
    v16bf a  = load_a_frag(aPtr + k);
    v16bf b0 = load_b_frag(bPtr + k);
    v16bf b1 = load_b_frag(bPtr + bStride     + k);
    v16bf b2 = load_b_frag(bPtr + 2 * bStride + k);
    v16bf b3 = load_b_frag(bPtr + 3 * bStride + k);
    // stream-prefetch the A row ahead (global_prefetch_b8)
    __builtin_prefetch(aPtr + k + 128, 0, 1);
    acc[0] = __builtin_amdgcn_wmma_f32_16x16x32_bf16(false, a, false, b0, (short)0, acc[0], false, false);
    acc[1] = __builtin_amdgcn_wmma_f32_16x16x32_bf16(false, a, false, b1, (short)0, acc[1], false, false);
    acc[2] = __builtin_amdgcn_wmma_f32_16x16x32_bf16(false, a, false, b2, (short)0, acc[2], false, false);
    acc[3] = __builtin_amdgcn_wmma_f32_16x16x32_bf16(false, a, false, b3, (short)0, acc[3], false, false);
  }

  // C/D layout: VGPR i, lanes 0-15 -> M=i, lanes 16-31 -> M=8+i; N = laneLo.
  const int mRow = mTile + (hi << 3);
  #pragma unroll
  for (int s = 0; s < 4; ++s) {
    const int n  = nBase + s * 16 + laneLo;
    const float bb = bias ? bias[n] : 0.0f;
    float* cp = C + (size_t)mRow * N + n;
    v8f av = acc[s];
    #pragma unroll
    for (int i = 0; i < 8; ++i) cp[(size_t)i * N] = av[i] + bb;
  }
}

// dst[N,K] (bf16) = transpose(src[K,N] f32): weights become K-major for WMMA B.
__global__ void transpose_f2b(const float* __restrict__ src, __bf16* __restrict__ dst,
                              int K, int N)
{
  int idx = blockIdx.x * blockDim.x + threadIdx.x;
  if (idx >= K * N) return;
  int k = idx / N, n = idx - k * N;
  dst[(size_t)n * K + k] = (__bf16)src[idx];
}

__global__ void convert_f2b(const float* __restrict__ src, __bf16* __restrict__ dst, int n)
{
  int idx = blockIdx.x * blockDim.x + threadIdx.x;
  if (idx < n) dst[idx] = (__bf16)src[idx];
}

// A1[B*T, E] = embed_bf[tgt_ids]
__global__ void gather_emb(const int* __restrict__ ids, const __bf16* __restrict__ emb,
                           __bf16* __restrict__ A1)
{
  int idx = blockIdx.x * blockDim.x + threadIdx.x;        // over B*T*E
  int row = idx >> 8;                                     // E = 256
  int e   = idx & 255;
  A1[(size_t)idx] = emb[(size_t)ids[row] * Ec + e];
}

__device__ __forceinline__ float sigmf(float x) { return 1.0f / (1.0f + __expf(-x)); }

// One GRU time step (Keras reset_after=True, gate order z,r,h; mask carries h).
// x: [B,T,3H] f32 (precomputed x@k + b[0]); inner: [B,3H] f32 (h@rk + b[1]).
__global__ void gru_gate(const float* __restrict__ x, int t,
                         const float* __restrict__ inner,
                         const int* __restrict__ ids,
                         float* __restrict__ h, __bf16* __restrict__ hb,
                         __bf16* __restrict__ seq)
{
  int idx = blockIdx.x * blockDim.x + threadIdx.x;        // over B*H
  int b = idx >> 9;                                       // H = 512
  int j = idx & 511;
  const float* xg = x     + ((size_t)b * Tc + t) * H3c;
  const float* rg = inner + (size_t)b * H3c;
  float z  = sigmf(xg[j]        + rg[j]);
  float r  = sigmf(xg[Hc  + j]  + rg[Hc  + j]);
  float hh = tanhf(xg[2*Hc + j] + r * rg[2*Hc + j]);
  float hp = h[idx];
  float hn = z * hp + (1.0f - z) * hh;
  float ho = (ids[b * Tc + t] != 0) ? hn : hp;
  h[idx]  = ho;
  hb[idx] = (__bf16)ho;
  seq[((size_t)b * Tc + t) * Hc + j] = (__bf16)ho;
}

// Row softmax over V=128: one wave per row (wave32, 4 elems/lane).
__global__ void softmax128(const float* __restrict__ logits, float* __restrict__ out)
{
  int lane = threadIdx.x & 31;
  int row  = blockIdx.x * 8 + (threadIdx.x >> 5);
  const float4 v = ((const float4*)(logits + (size_t)row * Vc))[lane];
  float mx = fmaxf(fmaxf(v.x, v.y), fmaxf(v.z, v.w));
  #pragma unroll
  for (int off = 16; off; off >>= 1) mx = fmaxf(mx, __shfl_xor(mx, off, 32));
  float e0 = __expf(v.x - mx), e1 = __expf(v.y - mx);
  float e2 = __expf(v.z - mx), e3 = __expf(v.w - mx);
  float s = e0 + e1 + e2 + e3;
  #pragma unroll
  for (int off = 16; off; off >>= 1) s += __shfl_xor(s, off, 32);
  float inv = 1.0f / s;
  float4 o = { e0 * inv, e1 * inv, e2 * inv, e3 * inv };
  ((float4*)(out + (size_t)row * Vc))[lane] = o;
}

extern "C" void kernel_launch(void* const* d_in, const int* in_sizes, int n_in,
                              void* d_out, int out_size, void* d_ws, size_t ws_size,
                              hipStream_t stream) {
  const int*   tgt  = (const int*)  d_in[0];
  const float* enc  = (const float*)d_in[1];
  const float* Wi1  = (const float*)d_in[2];
  const float* Wi2  = (const float*)d_in[3];
  const float* embT = (const float*)d_in[4];
  const float* k1   = (const float*)d_in[5];
  const float* rk1  = (const float*)d_in[6];
  const float* b1   = (const float*)d_in[7];
  const float* k2   = (const float*)d_in[8];
  const float* rk2  = (const float*)d_in[9];
  const float* b2   = (const float*)d_in[10];
  const float* Wv   = (const float*)d_in[11];
  const float* bv   = (const float*)d_in[12];
  float* probs = (float*)d_out;
  (void)in_sizes; (void)n_in; (void)out_size; (void)ws_size;

  // ---- workspace carve-up (256B aligned) ----
  char* ws = (char*)d_ws;
  size_t off = 0;
  auto alloc = [&](size_t bytes) -> char* {
    char* p = ws + off;
    off = (off + bytes + 255) & ~(size_t)255;
    return p;
  };
  const size_t BT = (size_t)Bc * Tc;                       // 98304 rows
  __bf16* Wi1_t  = (__bf16*)alloc((size_t)Hc  * ENCc * 2); // [H,ENC]
  __bf16* Wi2_t  = (__bf16*)alloc((size_t)Hc  * ENCc * 2);
  __bf16* k1_t   = (__bf16*)alloc((size_t)H3c * Ec   * 2); // [3H,E]
  __bf16* rk1_t  = (__bf16*)alloc((size_t)H3c * Hc   * 2); // [3H,H]
  __bf16* k2_t   = (__bf16*)alloc((size_t)H3c * Hc   * 2);
  __bf16* rk2_t  = (__bf16*)alloc((size_t)H3c * Hc   * 2);
  __bf16* Wv_t   = (__bf16*)alloc((size_t)Vc  * Hc   * 2); // [V,H]
  __bf16* emb_bf = (__bf16*)alloc((size_t)Vc  * Ec   * 2);
  __bf16* enc_bf = (__bf16*)alloc((size_t)Bc  * ENCc * 2);
  // gathered embeddings (bf16) and logits (f32) are disjoint in time -> share
  char*   shared = alloc(BT * Vc * 4);                     // 50.3 MB (>= BT*Ec*2)
  __bf16* A1     = (__bf16*)shared;
  float*  logits = (float*)shared;
  float*  xbuf   = (float*)alloc(BT * H3c * 4);            // x1 then x2 (reused)
  __bf16* seq1   = (__bf16*)alloc(BT * Hc * 2);
  __bf16* seq2   = (__bf16*)alloc(BT * Hc * 2);
  float*  h1f    = (float*) alloc((size_t)Bc * Hc * 4);
  float*  h2f    = (float*) alloc((size_t)Bc * Hc * 4);
  __bf16* h1b    = (__bf16*)alloc((size_t)Bc * Hc * 2);
  __bf16* h2b    = (__bf16*)alloc((size_t)Bc * Hc * 2);
  float*  inner  = (float*) alloc((size_t)Bc * H3c * 4);

  const dim3 blk(256);
  auto cdiv = [](long a, long b) { return (unsigned)((a + b - 1) / b); };

  // ---- weight / activation precision prep ----
  transpose_f2b<<<cdiv((long)ENCc * Hc, 256), blk, 0, stream>>>(Wi1, Wi1_t, ENCc, Hc);
  transpose_f2b<<<cdiv((long)ENCc * Hc, 256), blk, 0, stream>>>(Wi2, Wi2_t, ENCc, Hc);
  transpose_f2b<<<cdiv((long)Ec * H3c, 256), blk, 0, stream>>>(k1,  k1_t,  Ec, H3c);
  transpose_f2b<<<cdiv((long)Hc * H3c, 256), blk, 0, stream>>>(rk1, rk1_t, Hc, H3c);
  transpose_f2b<<<cdiv((long)Hc * H3c, 256), blk, 0, stream>>>(k2,  k2_t,  Hc, H3c);
  transpose_f2b<<<cdiv((long)Hc * H3c, 256), blk, 0, stream>>>(rk2, rk2_t, Hc, H3c);
  transpose_f2b<<<cdiv((long)Hc * Vc, 256), blk, 0, stream>>>(Wv, Wv_t, Hc, Vc);
  convert_f2b<<<cdiv((long)Vc * Ec, 256), blk, 0, stream>>>(embT, emb_bf, Vc * Ec);
  convert_f2b<<<cdiv((long)Bc * ENCc, 256), blk, 0, stream>>>(enc, enc_bf, Bc * ENCc);

  // ---- embedding gather + x1 = emb @ k1 + b1[0] ----
  gather_emb<<<cdiv((long)BT * Ec, 256), blk, 0, stream>>>(tgt, emb_bf, A1);
  gemm_bf16_nt<<<dim3(H3c / 128, (unsigned)(BT / 64)), blk, 0, stream>>>(
      A1, k1_t, b1, xbuf, (int)BT, H3c, Ec);

  // ---- initial states h0 = enc_hid @ W_init ----
  gemm_bf16_nt<<<dim3(Hc / 128, Bc / 64), blk, 0, stream>>>(
      enc_bf, Wi1_t, nullptr, h1f, Bc, Hc, ENCc);
  gemm_bf16_nt<<<dim3(Hc / 128, Bc / 64), blk, 0, stream>>>(
      enc_bf, Wi2_t, nullptr, h2f, Bc, Hc, ENCc);
  convert_f2b<<<cdiv((long)Bc * Hc, 256), blk, 0, stream>>>(h1f, h1b, Bc * Hc);
  convert_f2b<<<cdiv((long)Bc * Hc, 256), blk, 0, stream>>>(h2f, h2b, Bc * Hc);

  // ---- GRU layer 1: 48 recurrent steps ----
  for (int t = 0; t < Tc; ++t) {
    gemm_bf16_nt<<<dim3(H3c / 128, Bc / 64), blk, 0, stream>>>(
        h1b, rk1_t, b1 + H3c, inner, Bc, H3c, Hc);
    gru_gate<<<cdiv((long)Bc * Hc, 256), blk, 0, stream>>>(
        xbuf, t, inner, tgt, h1f, h1b, seq1);
  }

  // ---- x2 = seq1 @ k2 + b2[0] (reuse xbuf) ----
  gemm_bf16_nt<<<dim3(H3c / 128, (unsigned)(BT / 64)), blk, 0, stream>>>(
      seq1, k2_t, b2, xbuf, (int)BT, H3c, Hc);

  // ---- GRU layer 2 ----
  for (int t = 0; t < Tc; ++t) {
    gemm_bf16_nt<<<dim3(H3c / 128, Bc / 64), blk, 0, stream>>>(
        h2b, rk2_t, b2 + H3c, inner, Bc, H3c, Hc);
    gru_gate<<<cdiv((long)Bc * Hc, 256), blk, 0, stream>>>(
        xbuf, t, inner, tgt, h2f, h2b, seq2);
  }

  // ---- logits = seq2 @ Wv + bv; softmax over V=128 ----
  gemm_bf16_nt<<<dim3(Vc / 128, (unsigned)(BT / 64)), blk, 0, stream>>>(
      seq2, Wv_t, bv, logits, (int)BT, Vc, Hc);
  softmax128<<<cdiv((long)BT, 8), blk, 0, stream>>>(logits, probs);
}
